// GNNTrafficPredictor_44272522887550
// MI455X (gfx1250) — matrix operand order repaired and checked by reading, not compiled, not run on hardware
//
#include <hip/hip_runtime.h>
#include <hip/hip_bf16.h>

// ---------------------------------------------------------------------------
// GNN traffic predictor for MI455X (gfx1250, wave32, WMMA, async-to-LDS).
// ---------------------------------------------------------------------------

#define NN   4096          // nodes
#define EE   131072        // edges (w/o self loops)
#define ET   (EE + NN)     // edges incl self loops = 135168
#define HH   8             // heads
#define FF   64            // head dim
#define DD   512           // model dim
#define ND   (NN * DD)
#define NH   (NN * HH)

typedef __attribute__((ext_vector_type(16))) __bf16 v16bf;
typedef __attribute__((ext_vector_type(8)))  __bf16 v8bf;
typedef __attribute__((ext_vector_type(8)))  float  v8f;
typedef int v4i_ __attribute__((vector_size(16)));   // 16B chunk for async LDS

#if defined(__gfx1250__) && __has_builtin(__builtin_amdgcn_global_load_async_to_lds_b128)
#define USE_ASYNC_LDS 1
#else
#define USE_ASYNC_LDS 0
#endif

__device__ __forceinline__ v8f wmma_bf16(v16bf a, v16bf b, v8f c) {
  return __builtin_amdgcn_wmma_f32_16x16x32_bf16(
      false, a, false, b, (short)0, c, false, false);
}

__device__ __forceinline__ void atomicMaxF(float* addr, float val) {
  if (val >= 0.0f) atomicMax((int*)addr, __float_as_int(val));
  else             atomicMin((unsigned int*)addr, (unsigned int)__float_as_int(val));
}

// ---------------------------------------------------------------------------
// f32 -> bf16 transpose-convert for weights: out[Nc][K] = (bf16) in[K][Nc]
// ---------------------------------------------------------------------------
__global__ __launch_bounds__(256)
void cvt_bf16_T(const float* __restrict__ in, __bf16* __restrict__ out,
                int K, int Nc)
{
  int tid = blockIdx.x * 256 + threadIdx.x;      // K*Nc threads
  int k = tid / Nc, n = tid % Nc;                // coalesced read over n
  out[(size_t)n * K + k] = (__bf16)in[tid];
}

// ---------------------------------------------------------------------------
// Layer-0 projection: h[N,512] = x[N,6] @ W0[6,512]   (K=6: memory bound)
// ---------------------------------------------------------------------------
__global__ __launch_bounds__(256)
void gat0_proj(const float* __restrict__ x, const float* __restrict__ W0,
               float* __restrict__ h)
{
  int tid = blockIdx.x * 256 + threadIdx.x;      // ND threads
  int n = tid >> 9, c = tid & (DD - 1);
  const float* xr = x + (size_t)n * 6;
  float s = 0.f;
#pragma unroll
  for (int k = 0; k < 6; ++k) s += xr[k] * W0[k * DD + c];
  h[tid] = s;
}

// ---------------------------------------------------------------------------
// GEMM: C[M=4096,Nc] = act(A @ B + bias). A: bf16 [M][K] row-major.
// Bt: bf16 [Nc][K] (pre-transposed). K multiple of 32 (no guards).
// Template: HAS_BIAS, ACT (relu), OUT: 0=f32, 1=bf16, 2=bf16 transposed [Nc][M].
// Tile 128x64x32, 8 waves x (32x32). Double-buffered async global->LDS DMA.
// ---------------------------------------------------------------------------
#define TM 128
#define TN 64
#define TK 32
#define LSA 40
#define LSB 40

#define OUT_F32   0
#define OUT_BF16  1
#define OUT_BF16T 2

template <bool HAS_BIAS, bool ACT, int OUT>
__global__ __launch_bounds__(256)
void gemm_wmma(const __bf16* __restrict__ A, const __bf16* __restrict__ Bt,
               const float* __restrict__ bias, float* __restrict__ Cf,
               __bf16* __restrict__ Cb, int Nc, int K)
{
  __shared__ __bf16 lA[2 * TM * LSA];
  __shared__ __bf16 lB[2 * TN * LSB];

  const int t    = threadIdx.x;
  const int lane = t & 31;
  const int wave = t >> 5;
  const int wm   = wave >> 1;
  const int wn   = wave & 1;
  const int half = lane >> 4;
  const int l16  = lane & 15;
  const int brow = blockIdx.y * TM;
  const int bcol = blockIdx.x * TN;

  // A tile = 512 chunks of 8 halves: chunk c -> row c>>2, kg (c&3)*8
  const int ar0 = t >> 2, akg = (t & 3) << 3;    // chunks t and t+256
  const int ar1 = ar0 + 64;
  // B tile = 256 chunks: chunk t -> n t>>2, kg (t&3)*8
  const int bn_ = t >> 2, bkg = (t & 3) << 3;

  v8f acc[2][2] = {};
  const int kTiles = K / TK;

  auto stage = [&](int kt, int buf) {
    const int k0 = kt * TK;
    __bf16* dA = lA + buf * (TM * LSA);
    __bf16* dB = lB + buf * (TN * LSB);
#if USE_ASYNC_LDS
    __builtin_amdgcn_global_load_async_to_lds_b128(
        (v4i_*)(A + (size_t)(brow + ar0) * K + k0 + akg),
        (v4i_*)&dA[ar0 * LSA + akg], 0, 0);
    __builtin_amdgcn_global_load_async_to_lds_b128(
        (v4i_*)(A + (size_t)(brow + ar1) * K + k0 + akg),
        (v4i_*)&dA[ar1 * LSA + akg], 0, 0);
    __builtin_amdgcn_global_load_async_to_lds_b128(
        (v4i_*)(Bt + (size_t)(bcol + bn_) * K + k0 + bkg),
        (v4i_*)&dB[bn_ * LSB + bkg], 0, 0);
#else
    *(v8bf*)&dA[ar0 * LSA + akg] = *(const v8bf*)(A + (size_t)(brow + ar0) * K + k0 + akg);
    *(v8bf*)&dA[ar1 * LSA + akg] = *(const v8bf*)(A + (size_t)(brow + ar1) * K + k0 + akg);
    *(v8bf*)&dB[bn_ * LSB + bkg] = *(const v8bf*)(Bt + (size_t)(bcol + bn_) * K + k0 + bkg);
#endif
  };

  stage(0, 0);
  for (int kt = 0; kt < kTiles; ++kt) {
    const int cur = kt & 1;
    if (kt + 1 < kTiles) {
      stage(kt + 1, cur ^ 1);               // overlap next tile's DMA
#if USE_ASYNC_LDS
      asm volatile("s_wait_asynccnt 3" ::: "memory");   // oldest batch landed
#endif
    } else {
#if USE_ASYNC_LDS
      asm volatile("s_wait_asynccnt 0" ::: "memory");
#endif
    }
    __syncthreads();

    const __bf16* bufA = lA + cur * (TM * LSA);
    const __bf16* bufB = lB + cur * (TN * LSB);
    v16bf aF[2], bF[2];
#pragma unroll
    for (int s = 0; s < 2; ++s) {
      const int mb = wm * 32 + s * 16 + l16;
      const int nb = wn * 32 + s * 16 + l16;
#pragma unroll
      for (int j = 0; j < 16; ++j) {
        int kk = (half << 4) + j;                          // A 16x32 layout
        aF[s][j] = bufA[mb * LSA + kk];
        bF[s][j] = bufB[nb * LSB + (half << 4) + j];       // B 32x16 layout
      }
    }
#pragma unroll
    for (int sm = 0; sm < 2; ++sm)
#pragma unroll
      for (int sn = 0; sn < 2; ++sn)
        acc[sm][sn] = wmma_bf16(aF[sm], bF[sn], acc[sm][sn]);
    __syncthreads();
  }

  // ---- straight-line epilogue ----
  float bb[2] = {0.f, 0.f};
  if constexpr (HAS_BIAS) {
    bb[0] = bias[bcol + wn * 32 + l16];
    bb[1] = bias[bcol + wn * 32 + 16 + l16];
  }
#pragma unroll
  for (int sm = 0; sm < 2; ++sm)
#pragma unroll
    for (int sn = 0; sn < 2; ++sn) {
      const int col  = bcol + wn * 32 + sn * 16 + l16;
      if constexpr (OUT == OUT_BF16T) {
        const int rowb = brow + wm * 32 + sm * 16 + (half << 3);
        v8bf pk;
#pragma unroll
        for (int g = 0; g < 8; ++g) {
          float v = acc[sm][sn][g] + bb[sn];
          if constexpr (ACT) v = fmaxf(v, 0.0f);
          pk[g] = (__bf16)v;
        }
        *(v8bf*)&Cb[(size_t)col * NN + rowb] = pk;          // 16B store
      } else {
#pragma unroll
        for (int g = 0; g < 8; ++g) {
          const int row = brow + wm * 32 + sm * 16 + g + (half << 3);
          float v = acc[sm][sn][g] + bb[sn];
          if constexpr (ACT) v = fmaxf(v, 0.0f);
          if constexpr (OUT == OUT_F32) Cf[(size_t)row * Nc + col] = v;
          else                          Cb[(size_t)row * Nc + col] = (__bf16)v;
        }
      }
    }
}

// ---------------------------------------------------------------------------
// GAT edge phase
// ---------------------------------------------------------------------------
__global__ __launch_bounds__(256)
void gat_attn_sums(const float* __restrict__ h, const float* __restrict__ a_s,
                   const float* __restrict__ a_d, float* __restrict__ ssum,
                   float* __restrict__ dsum)
{
  int tid = blockIdx.x * 256 + threadIdx.x;      // NH threads
  int n = tid >> 3, hh = tid & 7;
  const float* hp  = h + (size_t)n * DD + hh * FF;
  const float* asp = a_s + hh * FF;
  const float* adp = a_d + hh * FF;
  float s1 = 0.f, s2 = 0.f;
#pragma unroll 8
  for (int f = 0; f < FF; ++f) { float v = hp[f]; s1 += v * asp[f]; s2 += v * adp[f]; }
  ssum[tid] = s1; dsum[tid] = s2;
}

__global__ __launch_bounds__(256)
void init_agg(float* __restrict__ agg, const float* __restrict__ bias)
{
  int tid = blockIdx.x * 256 + threadIdx.x;      // ND threads
  agg[tid] = bias[tid & (DD - 1)];
}

__global__ __launch_bounds__(256)
void init_seg(float* __restrict__ segm, float* __restrict__ segd)
{
  int tid = blockIdx.x * 256 + threadIdx.x;      // NH threads
  segm[tid] = -__builtin_inff();
  segd[tid] = 0.0f;
}

__device__ __forceinline__ void edge_endpoints(const int* ei, int e, int& src, int& dst)
{
  if (e < EE) { src = ei[e]; dst = ei[EE + e]; }
  else        { src = e - EE; dst = e - EE; }    // self loop
}

__global__ __launch_bounds__(256)
void edge_logits(const int* __restrict__ ei, const float* __restrict__ ssum,
                 const float* __restrict__ dsum, float* __restrict__ ebuf,
                 float* __restrict__ segm)
{
  int tid = blockIdx.x * 256 + threadIdx.x;      // ET*HH threads
  int e = tid >> 3, hh = tid & 7;
  int src, dst; edge_endpoints(ei, e, src, dst);
  float v = ssum[src * HH + hh] + dsum[dst * HH + hh];
  v = (v >= 0.f) ? v : 0.2f * v;                 // leaky_relu(0.2)
  ebuf[tid] = v;
  atomicMaxF(&segm[dst * HH + hh], v);
}

__global__ __launch_bounds__(256)
void edge_exp(const int* __restrict__ ei, float* __restrict__ ebuf,
              const float* __restrict__ segm, float* __restrict__ segd)
{
  int tid = blockIdx.x * 256 + threadIdx.x;
  int e = tid >> 3, hh = tid & 7;
  int src, dst; edge_endpoints(ei, e, src, dst);
  (void)src;
  float ex = __expf(ebuf[tid] - segm[dst * HH + hh]);
  ebuf[tid] = ex;
  atomicAdd(&segd[dst * HH + hh], ex);
}

__global__ __launch_bounds__(256)
void edge_scatter(const int* __restrict__ ei, const float* __restrict__ ebuf,
                  const float* __restrict__ segd, const float* __restrict__ h,
                  float* __restrict__ agg)
{
  int e = blockIdx.x;                            // ET blocks
  int src, dst; edge_endpoints(ei, e, src, dst);
  int t = threadIdx.x;
#pragma unroll
  for (int rep = 0; rep < 2; ++rep) {
    int d = t + rep * 256;
    int hh = d >> 6;
    float alpha = ebuf[e * HH + hh] / (segd[dst * HH + hh] + 1e-16f);
    atomicAdd(&agg[(size_t)dst * DD + d], h[(size_t)src * DD + d] * alpha);
  }
}

// ---------------------------------------------------------------------------
// BatchNorm (biased var) + ReLU -> bf16 activations
// ---------------------------------------------------------------------------
__global__ __launch_bounds__(256)
void bn_stats(const float* __restrict__ xin, float* __restrict__ stat)
{
  int c = blockIdx.x, t = threadIdx.x;
  float s = 0.f, q = 0.f;
  for (int r = t; r < NN; r += 256) {
    float v = xin[(size_t)r * DD + c];
    s += v; q += v * v;
  }
  __shared__ float sh[256], shq[256];
  sh[t] = s; shq[t] = q; __syncthreads();
  for (int o = 128; o > 0; o >>= 1) {
    if (t < o) { sh[t] += sh[t + o]; shq[t] += shq[t + o]; }
    __syncthreads();
  }
  if (t == 0) {
    float mu = sh[0] / (float)NN;
    stat[c] = mu;
    stat[DD + c] = shq[0] / (float)NN - mu * mu;
  }
}

__global__ __launch_bounds__(256)
void bn_apply_relu(const float* __restrict__ xin, const float* __restrict__ stat,
                   const float* __restrict__ g, const float* __restrict__ b,
                   __bf16* __restrict__ out)
{
  int tid = blockIdx.x * 256 + threadIdx.x;      // ND threads
  int c = tid & (DD - 1);
  float mu = stat[c], var = stat[DD + c];
  float v = (xin[tid] - mu) * rsqrtf(var + 1e-5f) * g[c] + b[c];
  out[tid] = (__bf16)fmaxf(v, 0.0f);
}

// ---------------------------------------------------------------------------
// Flash attention: one wave per (head, 16-query tile); 32-key chunks.
// Q,K bf16 row-major [N][D]; Vt bf16 [D][N] (pre-transposed) -> contiguous
// B-fragment reads everywhere. Output ao in bf16 (next GEMM's A operand).
// ---------------------------------------------------------------------------
__global__ __launch_bounds__(256)
void flash_attn(const __bf16* __restrict__ Q, const __bf16* __restrict__ Kb,
                const __bf16* __restrict__ Vt, __bf16* __restrict__ O)
{
  __shared__ __bf16 plds[8 * 16 * 40];

  const int lane = threadIdx.x & 31;
  const int wave = threadIdx.x >> 5;
  const int wid  = blockIdx.x * 8 + wave;        // 2048 waves
  const int head = wid >> 8;
  const int qt   = wid & 255;
  const int half = lane >> 4;
  const int l16  = lane & 15;
  __bf16* myP = &plds[wave * 16 * 40];

  const float scale = 0.125f;                    // 1/sqrt(64)
  const int qrow = qt * 16 + l16;

  v16bf aq0, aq1;
#pragma unroll
  for (int j = 0; j < 16; ++j) {
    int k0 = (half << 4) + j;
    aq0[j] = Q[(size_t)qrow * DD + head * FF + k0];
    aq1[j] = Q[(size_t)qrow * DD + head * FF + 32 + k0];
  }

  float mrow[8], srow[8];
  v8f o0 = {}, o1 = {}, o2 = {}, o3 = {};
#pragma unroll
  for (int g = 0; g < 8; ++g) { mrow[g] = -__builtin_inff(); srow[g] = 0.f; }

  for (int kc = 0; kc < NN; kc += 32) {
    v16bf bk00, bk01, bk10, bk11;
    const int key0 = kc + l16, key1 = kc + 16 + l16;
    const int fb = (half << 4);
#pragma unroll
    for (int j = 0; j < 16; ++j) {
      bk00[j] = Kb[(size_t)key0 * DD + head * FF + fb + j];
      bk01[j] = Kb[(size_t)key0 * DD + head * FF + 32 + fb + j];
      bk10[j] = Kb[(size_t)key1 * DD + head * FF + fb + j];
      bk11[j] = Kb[(size_t)key1 * DD + head * FF + 32 + fb + j];
    }
    v8f s0 = {}, s1 = {};
    s0 = wmma_bf16(aq0, bk00, s0); s0 = wmma_bf16(aq1, bk01, s0);
    s1 = wmma_bf16(aq0, bk10, s1); s1 = wmma_bf16(aq1, bk11, s1);

#pragma unroll
    for (int g = 0; g < 8; ++g) {
      float a = s0[g] * scale, b = s1[g] * scale;
      float v = fmaxf(a, b);
#pragma unroll
      for (int m = 1; m <= 8; m <<= 1) v = fmaxf(v, __shfl_xor(v, m, 32));
      float mnew = fmaxf(mrow[g], v);
      float p0 = __expf(a - mnew), p1 = __expf(b - mnew);
      float rs = p0 + p1;
#pragma unroll
      for (int m = 1; m <= 8; m <<= 1) rs += __shfl_xor(rs, m, 32);
      float corr = __expf(mrow[g] - mnew);
      srow[g] = srow[g] * corr + rs;
      mrow[g] = mnew;
      o0[g] *= corr; o1[g] *= corr; o2[g] *= corr; o3[g] *= corr;
      int r = g + (half << 3);
      myP[r * 40 + l16]      = (__bf16)p0;
      myP[r * 40 + 16 + l16] = (__bf16)p1;
    }

    v16bf ap;
#pragma unroll
    for (int j = 0; j < 16; ++j) {
      int kk = (half << 4) + j;
      ap[j] = myP[l16 * 40 + kk];
    }

#pragma unroll
    for (int d0 = 0; d0 < 4; ++d0) {
      v16bf bv;
#pragma unroll
      for (int j = 0; j < 16; ++j)
        bv[j] = Vt[(size_t)(head * FF + d0 * 16 + l16) * NN + kc + (half << 4) + j];
      if      (d0 == 0) o0 = wmma_bf16(ap, bv, o0);
      else if (d0 == 1) o1 = wmma_bf16(ap, bv, o1);
      else if (d0 == 2) o2 = wmma_bf16(ap, bv, o2);
      else              o3 = wmma_bf16(ap, bv, o3);
    }
  }

#pragma unroll
  for (int g = 0; g < 8; ++g) {
    int row = qt * 16 + g + (half << 3);
    float inv = 1.0f / srow[g];
    size_t base = (size_t)row * DD + head * FF;
    O[base + l16]      = (__bf16)(o0[g] * inv);
    O[base + 16 + l16] = (__bf16)(o1[g] * inv);
    O[base + 32 + l16] = (__bf16)(o2[g] * inv);
    O[base + 48 + l16] = (__bf16)(o3[g] * inv);
  }
}

// ---------------------------------------------------------------------------
// Final: out[n] = h2[n,:128] . Wm2 + bm2 (one wave per row)
// ---------------------------------------------------------------------------
__global__ __launch_bounds__(256)
void final_dot(const __bf16* __restrict__ h2, const float* __restrict__ w,
               const float* __restrict__ b, float* __restrict__ out)
{
  int row  = blockIdx.x * 8 + (threadIdx.x >> 5);
  int lane = threadIdx.x & 31;
  const __bf16* hp = h2 + (size_t)row * 128;
  float s = 0.f;
#pragma unroll
  for (int i = lane; i < 128; i += 32) s += (float)hp[i] * w[i];
#pragma unroll
  for (int m = 16; m > 0; m >>= 1) s += __shfl_xor(s, m, 32);
  if (lane == 0) out[row] = s + b[0];
}

// ---------------------------------------------------------------------------
// Host launch
// ---------------------------------------------------------------------------
template <bool HAS_BIAS, bool ACT, int OUT>
static inline void launch_gemm(const __bf16* A, const __bf16* Bt, const float* bias,
                               float* Cf, __bf16* Cb, int Nc, int K, hipStream_t s)
{
  dim3 grid(Nc / TN, NN / TM);
  gemm_wmma<HAS_BIAS, ACT, OUT><<<grid, 256, 0, s>>>(A, Bt, bias, Cf, Cb, Nc, K);
}

extern "C" void kernel_launch(void* const* d_in, const int* in_sizes, int n_in,
                              void* d_out, int out_size, void* d_ws, size_t ws_size,
                              hipStream_t stream)
{
  (void)in_sizes; (void)n_in; (void)out_size; (void)ws_size;

  const float* x  = (const float*)d_in[0];
  const int*   ei = (const int*)d_in[1];
  const float* Wl[3]  = {(const float*)d_in[2], (const float*)d_in[6],  (const float*)d_in[10]};
  const float* asl[3] = {(const float*)d_in[3], (const float*)d_in[7],  (const float*)d_in[11]};
  const float* adl[3] = {(const float*)d_in[4], (const float*)d_in[8],  (const float*)d_in[12]};
  const float* bl[3]  = {(const float*)d_in[5], (const float*)d_in[9],  (const float*)d_in[13]};
  const float* bn_g = (const float*)d_in[14];
  const float* bn_b = (const float*)d_in[15];
  const float* Wq = (const float*)d_in[16]; const float* bq = (const float*)d_in[17];
  const float* Wk = (const float*)d_in[18]; const float* bk = (const float*)d_in[19];
  const float* Wv = (const float*)d_in[20]; const float* bv = (const float*)d_in[21];
  const float* Wo = (const float*)d_in[22]; const float* bo = (const float*)d_in[23];
  const float* Wm0 = (const float*)d_in[24]; const float* bm0 = (const float*)d_in[25];
  const float* Wm1 = (const float*)d_in[26]; const float* bm1 = (const float*)d_in[27];
  const float* Wm2 = (const float*)d_in[28]; const float* bm2 = (const float*)d_in[29];

  // ---- workspace layout ----
  char* p = (char*)d_ws;
  float* F0   = (float*)p; p += (size_t)ND * 4;          // h (GAT projection)
  float* F1   = (float*)p; p += (size_t)ND * 4;          // agg
  float* ebuf = (float*)p; p += (size_t)ET * HH * 4;
  float* ssum = (float*)p; p += (size_t)NH * 4;
  float* dsum = (float*)p; p += (size_t)NH * 4;
  float* segm = (float*)p; p += (size_t)NH * 4;
  float* segd = (float*)p; p += (size_t)NH * 4;
  float* stat = (float*)p; p += 2 * DD * 4;
  __bf16* X0  = (__bf16*)p; p += (size_t)ND * 2;         // xcur / ao
  __bf16* X1  = (__bf16*)p; p += (size_t)ND * 2;         // q / Wo-out
  __bf16* X2  = (__bf16*)p; p += (size_t)ND * 2;         // k / h1
  __bf16* X3  = (__bf16*)p; p += (size_t)ND * 2;         // vT / h2
  __bf16* Btb = (__bf16*)p;                              // transposed weight (<=512x512)

  // ---- 3 GAT layers ----
  for (int i = 0; i < 3; ++i) {
    if (i == 0) {
      gat0_proj<<<ND / 256, 256, 0, stream>>>(x, Wl[0], F0);
    } else {
      cvt_bf16_T<<<(DD * DD) / 256, 256, 0, stream>>>(Wl[i], Btb, DD, DD);
      launch_gemm<false, false, OUT_F32>(X0, Btb, nullptr, F0, nullptr, DD, DD, stream);
    }
    gat_attn_sums<<<NH / 256, 256, 0, stream>>>(F0, asl[i], adl[i], ssum, dsum);
    init_agg<<<ND / 256, 256, 0, stream>>>(F1, bl[i]);
    init_seg<<<NH / 256, 256, 0, stream>>>(segm, segd);
    edge_logits<<<(ET * HH) / 256, 256, 0, stream>>>(ei, ssum, dsum, ebuf, segm);
    edge_exp<<<(ET * HH) / 256, 256, 0, stream>>>(ei, ebuf, segm, segd);
    edge_scatter<<<ET, 256, 0, stream>>>(ei, ebuf, segd, F0, F1);
    bn_stats<<<DD, 256, 0, stream>>>(F1, stat);
    bn_apply_relu<<<ND / 256, 256, 0, stream>>>(F1, stat, bn_g + i * DD, bn_b + i * DD, X0);
  }

  // ---- self-attention (q,k row-major; v transposed for flash PV) ----
  cvt_bf16_T<<<(DD * DD) / 256, 256, 0, stream>>>(Wq, Btb, DD, DD);
  launch_gemm<true, false, OUT_BF16>(X0, Btb, bq, nullptr, X1, DD, DD, stream);
  cvt_bf16_T<<<(DD * DD) / 256, 256, 0, stream>>>(Wk, Btb, DD, DD);
  launch_gemm<true, false, OUT_BF16>(X0, Btb, bk, nullptr, X2, DD, DD, stream);
  cvt_bf16_T<<<(DD * DD) / 256, 256, 0, stream>>>(Wv, Btb, DD, DD);
  launch_gemm<true, false, OUT_BF16T>(X0, Btb, bv, nullptr, X3, DD, DD, stream); // Vt [D][N]
  flash_attn<<<256, 256, 0, stream>>>(X1, X2, X3, X0);           // ao -> X0 (bf16)
  cvt_bf16_T<<<(DD * DD) / 256, 256, 0, stream>>>(Wo, Btb, DD, DD);
  launch_gemm<true, false, OUT_BF16>(X0, Btb, bo, nullptr, X1, DD, DD, stream);

  // ---- MLP (bf16 end-to-end) ----
  cvt_bf16_T<<<(DD * DD / 2) / 256, 256, 0, stream>>>(Wm0, Btb, DD, DD / 2);
  launch_gemm<true, true, OUT_BF16>(X1, Btb, bm0, nullptr, X2, DD / 2, DD, stream);
  cvt_bf16_T<<<(DD / 2 * DD / 4) / 256, 256, 0, stream>>>(Wm1, Btb, DD / 2, DD / 4);
  launch_gemm<true, true, OUT_BF16>(X2, Btb, bm1, nullptr, X3, DD / 4, DD / 2, stream);
  final_dot<<<NN / 8, 256, 0, stream>>>(X3, Wm2, bm2, (float*)d_out);
}
